// Model_43671227465813
// MI455X (gfx1250) — matrix operand order repaired and checked by reading, not compiled, not run on hardware
//
#include <hip/hip_runtime.h>

#define N_NODES   50000
#define BAG_LEN   32
#define HID       128
#define N_CLASSES 16
#define N_EDGES   1600000
#define N_LAYERS  5

#define GEMM_BLOCKS 256          // 256 blocks x 8 waves = 2048 persistent waves
#define GEMM_WAVES  (GEMM_BLOCKS * 8)
#define ROW_TILES   (N_NODES / 16)   // 3125

typedef __attribute__((ext_vector_type(16))) __bf16 v16bf;
typedef __attribute__((ext_vector_type(8)))  float  v8f;

struct __attribute__((aligned(16))) U4 { unsigned int x, y, z, w; };
union Frag16 { v16bf v; U4 q[2]; };

// ---------------------------------------------------------------------------
// Weight prep: Wconv[l][k][n] -> WT[l][n][k] bf16 ; Wlin[c][k] -> WL bf16
// ---------------------------------------------------------------------------
__global__ void k_prep_w(const float* __restrict__ Wconv, const float* __restrict__ Wlin,
                         __bf16* __restrict__ WT, __bf16* __restrict__ WL) {
  int i = blockIdx.x * blockDim.x + threadIdx.x;
  const int nconv = N_LAYERS * HID * HID;
  if (i < nconv) {
    int l = i / (HID * HID);
    int r = i - l * HID * HID;
    int n = r / HID;
    int k = r - n * HID;
    WT[i] = (__bf16)Wconv[(size_t)l * HID * HID + (size_t)k * HID + n];
  } else if (i < nconv + N_CLASSES * HID) {
    int j = i - nconv;
    WL[j] = (__bf16)Wlin[j];
  }
}

// ---------------------------------------------------------------------------
// GCN degree: deg = 1 (self loop) + indeg ; then in-place rsqrt
// ---------------------------------------------------------------------------
__global__ void k_deg_init(float* __restrict__ deg) {
  int i = blockIdx.x * blockDim.x + threadIdx.x;
  if (i < N_NODES) deg[i] = 1.0f;
}
__global__ void k_deg_count(const int* __restrict__ ei, float* __restrict__ deg) {
  int e = blockIdx.x * blockDim.x + threadIdx.x;
  if (e < N_EDGES) atomicAdd(&deg[ei[N_EDGES + e]], 1.0f);
}
__global__ void k_inv_sqrt(float* __restrict__ deg) {
  int i = blockIdx.x * blockDim.x + threadIdx.x;
  if (i < N_NODES) deg[i] = rsqrtf(deg[i]);
}

// ---------------------------------------------------------------------------
// EmbeddingBag mean + ReLU -> bf16 node features. One wave per node,
// each lane owns 4 consecutive features (float4 gathers).
// ---------------------------------------------------------------------------
__global__ void k_embed(const int* __restrict__ x, const float* __restrict__ emb,
                        __bf16* __restrict__ hb) {
  const int wid  = (int)((blockIdx.x * blockDim.x + threadIdx.x) >> 5);
  const int lane = threadIdx.x & 31;
  if (wid >= N_NODES) return;
  const int* bag = x + (size_t)wid * BAG_LEN;
  const int col = lane * 4;
  float ax = 0.f, ay = 0.f, az = 0.f, aw = 0.f;
  for (int j = 0; j < BAG_LEN; ++j) {
    const float4 e = *(const float4*)(emb + (size_t)bag[j] * HID + col);
    ax += e.x; ay += e.y; az += e.z; aw += e.w;
  }
  const float s = 1.0f / (float)BAG_LEN;
  __bf16 p[4] = { (__bf16)fmaxf(ax * s, 0.f), (__bf16)fmaxf(ay * s, 0.f),
                  (__bf16)fmaxf(az * s, 0.f), (__bf16)fmaxf(aw * s, 0.f) };
  unsigned long long u; __builtin_memcpy(&u, p, 8);
  *(unsigned long long*)(hb + (size_t)wid * HID + col) = u;
}

// ---------------------------------------------------------------------------
// m = h @ W  via v_wmma_f32_16x16x32_bf16, persistent-wave version.
//
// Each wave owns a fixed 32-column quarter and holds its 8 B fragments
// (4 k-steps x 2 n-tiles = 64 VGPRs) in registers for its whole lifetime
// (~6 row tiles), so the hot loop is only 8x global_load_b128 (A) +
// 8 WMMA + staging. 64 B-VGPRs + 16 acc + A keeps total pressure ~130,
// below the spill threshold that hurt the previous version.
// The 16x32 f32 result tile is staged through a private 2KB LDS slice
// (within-wave DS ordering, no barrier) to emit coalesced b128 stores.
//
// A (bf16 16x32): lane[0:15]=rows, contiguous K run kh*16+[0..15]
// B (bf16 32x16): lane holds column n=lane&15, contiguous K run kh*16+[0..15]
// C (f32 16x16):  lane&15 = N, VGPR r -> M = (lane>>4)*8 + r
// ---------------------------------------------------------------------------
__global__ void k_gcn_gemm(const __bf16* __restrict__ A,
                           const __bf16* __restrict__ BT,
                           float* __restrict__ C) {
  __shared__ float lds[8 * 16 * 32];            // 2KB per wave, 8 waves/block
  float* wlds = lds + (threadIdx.x >> 5) * (16 * 32);

  const int gwid = (int)((blockIdx.x * blockDim.x + threadIdx.x) >> 5);
  const int lane = threadIdx.x & 31;
  const int ml = lane & 15;
  const int kh = lane >> 4;
  const int nq = gwid & 3;                      // column quarter
  const int nbase = nq * 2;                     // first of 2 n-tiles
  const int colbase = nq * 32;

  // ---- preload the 8 B fragments for this wave's column quarter ----
  Frag16 bf[4][2];
#pragma unroll
  for (int kk = 0; kk < 4; ++kk) {
    const int kb = kk * 32 + kh * 16;
#pragma unroll
    for (int nt = 0; nt < 2; ++nt) {
      const __bf16* brow = BT + ((size_t)(nbase + nt) * 16 + ml) * HID + kb;
      bf[kk][nt].q[0] = *(const U4*)(brow);
      bf[kk][nt].q[1] = *(const U4*)(brow + 8);
    }
  }

  const v8f vzero = {0.f, 0.f, 0.f, 0.f, 0.f, 0.f, 0.f, 0.f};

  for (int tile = gwid >> 2; tile < ROW_TILES; tile += (GEMM_WAVES / 4)) {
    const __bf16* arow = A + ((size_t)tile * 16 + ml) * HID;

    v8f acc[2];
    acc[0] = vzero;
    acc[1] = vzero;

#pragma unroll
    for (int kk = 0; kk < 4; ++kk) {
      Frag16 a;
      const int ka = kk * 32 + kh * 16;
      a.q[0] = *(const U4*)(arow + ka);
      a.q[1] = *(const U4*)(arow + ka + 8);
#pragma unroll
      for (int nt = 0; nt < 2; ++nt) {
        acc[nt] = __builtin_amdgcn_wmma_f32_16x16x32_bf16(
            false, a.v, false, bf[kk][nt].v, (short)0, acc[nt], false, false);
      }
    }

    // ---- stage C tile into this wave's LDS slice (fragment layout) ----
#pragma unroll
    for (int nt = 0; nt < 2; ++nt) {
#pragma unroll
      for (int r = 0; r < 8; ++r)
        wlds[(kh * 8 + r) * 32 + nt * 16 + ml] = acc[nt][r];
    }
    // ---- read back row-major, emit coalesced b128 global stores ----
#pragma unroll
    for (int pass = 0; pass < 4; ++pass) {
      const int row = pass * 4 + (lane >> 3);
      const int c4  = (lane & 7) * 4;
      const float4 v = *(const float4*)(wlds + row * 32 + c4);
      *(float4*)(C + ((size_t)tile * 16 + row) * HID + colbase + c4) = v;
    }
  }
}

// ---------------------------------------------------------------------------
// agg init with self-loop term: agg[n] = m[n] * inv[n]^2
// ---------------------------------------------------------------------------
__global__ void k_self_init(const float* __restrict__ m, const float* __restrict__ inv,
                            float* __restrict__ agg) {
  int t = blockIdx.x * blockDim.x + threadIdx.x;
  if (t >= N_NODES * (HID / 4)) return;
  int node = t / (HID / 4);
  int c = (t - node * (HID / 4)) * 4;
  float s = inv[node]; s *= s;
  const float4 mv = *(const float4*)(m + (size_t)node * HID + c);
  float4 o; o.x = mv.x * s; o.y = mv.y * s; o.z = mv.z * s; o.w = mv.w * s;
  *(float4*)(agg + (size_t)node * HID + c) = o;
}

// ---------------------------------------------------------------------------
// edge scatter: agg[dst] += m[src] * inv[src]*inv[dst]. One wave per edge.
// ---------------------------------------------------------------------------
__global__ void k_scatter(const int* __restrict__ ei, const float* __restrict__ inv,
                          const float* __restrict__ m, float* __restrict__ agg) {
  const int wid  = (int)((blockIdx.x * blockDim.x + threadIdx.x) >> 5);
  const int lane = threadIdx.x & 31;
  if (wid >= N_EDGES) return;
  const int s = ei[wid];
  const int d = ei[N_EDGES + wid];
  const float nrm = inv[s] * inv[d];
  const int c = lane * 4;
  const float4 mv = *(const float4*)(m + (size_t)s * HID + c);
  float* ap = agg + (size_t)d * HID + c;
  atomicAdd(ap + 0, mv.x * nrm);
  atomicAdd(ap + 1, mv.y * nrm);
  atomicAdd(ap + 2, mv.z * nrm);
  atomicAdd(ap + 3, mv.w * nrm);
}

// ---------------------------------------------------------------------------
// h = relu(agg), re-quantize to bf16 for next WMMA GEMM
// ---------------------------------------------------------------------------
__global__ void k_relu_pack(const float* __restrict__ agg, __bf16* __restrict__ hb) {
  int t = blockIdx.x * blockDim.x + threadIdx.x;
  if (t >= N_NODES * (HID / 4)) return;
  const float4 v = *(const float4*)(agg + (size_t)t * 4);
  __bf16 p[4] = { (__bf16)fmaxf(v.x, 0.f), (__bf16)fmaxf(v.y, 0.f),
                  (__bf16)fmaxf(v.z, 0.f), (__bf16)fmaxf(v.w, 0.f) };
  unsigned long long u; __builtin_memcpy(&u, p, 8);
  *(unsigned long long*)(hb + (size_t)t * 4) = u;
}

// ---------------------------------------------------------------------------
// logits = h @ Wlin^T via WMMA. Wlin[c][k] row-major is already the [n][k]
// B-fragment layout. One wave per 16-row tile, single 16-col tile.
// ---------------------------------------------------------------------------
__global__ void k_linear(const __bf16* __restrict__ A, const __bf16* __restrict__ WL,
                         float* __restrict__ out) {
  const int wid  = (int)((blockIdx.x * blockDim.x + threadIdx.x) >> 5);
  const int lane = threadIdx.x & 31;
  if (wid >= ROW_TILES) return;
  const int ml = lane & 15;
  const int kh = lane >> 4;
  const __bf16* arow = A + ((size_t)wid * 16 + ml) * HID;
  const __bf16* brow = WL + (size_t)ml * HID;
  v8f acc = {0.f, 0.f, 0.f, 0.f, 0.f, 0.f, 0.f, 0.f};
#pragma unroll
  for (int kk = 0; kk < 4; ++kk) {
    Frag16 a, b;
    const int ka = kk * 32 + kh * 16;
    a.q[0] = *(const U4*)(arow + ka);
    a.q[1] = *(const U4*)(arow + ka + 8);
    const int kb = kk * 32 + kh * 16;
    b.q[0] = *(const U4*)(brow + kb);
    b.q[1] = *(const U4*)(brow + kb + 8);
    acc = __builtin_amdgcn_wmma_f32_16x16x32_bf16(
        false, a.v, false, b.v, (short)0, acc, false, false);
  }
#pragma unroll
  for (int r = 0; r < 8; ++r)
    out[((size_t)wid * 16 + kh * 8 + r) * N_CLASSES + ml] = acc[r];
}

// ---------------------------------------------------------------------------
// in-place row softmax over 16 classes
// ---------------------------------------------------------------------------
__global__ void k_softmax(float* __restrict__ out) {
  int n = blockIdx.x * blockDim.x + threadIdx.x;
  if (n >= N_NODES) return;
  float* row = out + (size_t)n * N_CLASSES;
  float v[N_CLASSES];
  float mx = -3.4e38f;
#pragma unroll
  for (int c = 0; c < N_CLASSES; ++c) { v[c] = row[c]; mx = fmaxf(mx, v[c]); }
  float sum = 0.f;
#pragma unroll
  for (int c = 0; c < N_CLASSES; ++c) { v[c] = __expf(v[c] - mx); sum += v[c]; }
  const float r = 1.f / sum;
#pragma unroll
  for (int c = 0; c < N_CLASSES; ++c) row[c] = v[c] * r;
}

// ---------------------------------------------------------------------------
extern "C" void kernel_launch(void* const* d_in, const int* in_sizes, int n_in,
                              void* d_out, int out_size, void* d_ws, size_t ws_size,
                              hipStream_t stream) {
  const int*   x     = (const int*)d_in[0];
  const int*   ei    = (const int*)d_in[1];
  const float* emb   = (const float*)d_in[2];
  const float* Wconv = (const float*)d_in[3];
  const float* Wlin  = (const float*)d_in[4];
  float* out = (float*)d_out;

  char* ws = (char*)d_ws;
  float*  m   = (float*)ws;   ws += (size_t)N_NODES * HID * sizeof(float);
  float*  agg = (float*)ws;   ws += (size_t)N_NODES * HID * sizeof(float);
  float*  deg = (float*)ws;   ws += (size_t)N_NODES * sizeof(float);
  __bf16* hb  = (__bf16*)ws;  ws += (size_t)N_NODES * HID * sizeof(__bf16);
  __bf16* WT  = (__bf16*)ws;  ws += (size_t)N_LAYERS * HID * HID * sizeof(__bf16);
  __bf16* WL  = (__bf16*)ws;

  k_prep_w<<<(N_LAYERS*HID*HID + N_CLASSES*HID + 255) / 256, 256, 0, stream>>>(
      Wconv, Wlin, WT, WL);
  k_deg_init<<<(N_NODES + 255) / 256, 256, 0, stream>>>(deg);
  k_deg_count<<<(N_EDGES + 255) / 256, 256, 0, stream>>>(ei, deg);
  k_inv_sqrt<<<(N_NODES + 255) / 256, 256, 0, stream>>>(deg);
  k_embed<<<(N_NODES * 32 + 255) / 256, 256, 0, stream>>>(x, emb, hb);

  for (int l = 0; l < N_LAYERS; ++l) {
    k_gcn_gemm<<<GEMM_BLOCKS, 256, 0, stream>>>(hb, WT + (size_t)l * HID * HID, m);
    k_self_init<<<(N_NODES * (HID / 4) + 255) / 256, 256, 0, stream>>>(m, deg, agg);
    k_scatter<<<(N_EDGES * 32 + 255) / 256, 256, 0, stream>>>(ei, deg, m, agg);
    k_relu_pack<<<(N_NODES * (HID / 4) + 255) / 256, 256, 0, stream>>>(agg, hb);
  }

  k_linear<<<((ROW_TILES) * 32 + 255) / 256, 256, 0, stream>>>(hb, WL, out);
  k_softmax<<<(N_NODES + 255) / 256, 256, 0, stream>>>(out);
}
